// ModalitySpecificLocalCrossAttention_30657476559025
// MI455X (gfx1250) — matrix operand order repaired and checked by reading, not compiled, not run on hardware
//
#include <hip/hip_runtime.h>

// ---------------------------------------------------------------------------
// ModalitySpecificLocalCrossAttention for MI455X (gfx1250, wave32, WMMA).
// Convs run as bf16 WMMA GEMMs (v_wmma_f32_16x16x32_bf16), f32 accumulate;
// 3x3 convs = 9 accumulated shifted GEMMs. A tiles staged into LDS with
// gfx1250 async copies (global_load_async_to_lds_b128 + s_wait_asynccnt).
// Local 9x9 attention: one wave per pixel, online softmax.
// ---------------------------------------------------------------------------

typedef __attribute__((ext_vector_type(16))) __bf16 v16bf;
typedef __attribute__((ext_vector_type(8)))  float  v8f;
typedef unsigned short ushort_t;
typedef __attribute__((address_space(3))) ushort_t lds_ushort;

static constexpr int NB_  = 2;
static constexpr int H_   = 180;
static constexpr int W_   = 180;
static constexpr int HW_  = H_ * W_;       // 32400
static constexpr int P_   = NB_ * HW_;     // 64800 pixels (divisible by 32)
static constexpr int MAX_LDA = 512;

__device__ __forceinline__ ushort_t f2bf(float f) {
  union { float f; unsigned u; } c; c.f = f;
  unsigned r = c.u + 0x7FFFu + ((c.u >> 16) & 1u);   // round-to-nearest-even
  return (ushort_t)(r >> 16);
}
__device__ __forceinline__ float bf2f(ushort_t u) {
  union { unsigned u; float f; } c; c.u = ((unsigned)u) << 16;
  return c.f;
}

// ---------------------------------------------------------------------------
// NCHW f32 -> pixel-major bf16 matrix [P][Cpad], zero-padding channels.
// ---------------------------------------------------------------------------
__global__ void k_tomat(const float* __restrict__ src, ushort_t* __restrict__ dst,
                        int C, int Cpad) {
  size_t i = (size_t)blockIdx.x * blockDim.x + threadIdx.x;
  size_t tot = (size_t)P_ * Cpad;
  if (i >= tot) return;
  int c = (int)(i % Cpad);
  size_t p = i / Cpad;
  float v = 0.f;
  if (c < C) {
    int n = (int)(p / HW_), hw = (int)(p % HW_);
    v = src[((size_t)n * C + c) * HW_ + hw];
  }
  dst[i] = f2bf(v);
}

// ---------------------------------------------------------------------------
// Weight (O,I,kh,kw) f32 -> bf16 Wt[t][o][Kpad] (B^T layout: K contiguous).
// Columns c >= I are zero-filled (absorbs activation channel padding).
// ---------------------------------------------------------------------------
__global__ void k_wconv(const float* __restrict__ w, ushort_t* __restrict__ dst,
                        int O, int I, int kh, int Kpad) {
  int taps = kh * kh;
  size_t i = (size_t)blockIdx.x * blockDim.x + threadIdx.x;
  size_t tot = (size_t)taps * O * Kpad;
  if (i >= tot) return;
  int c = (int)(i % Kpad);
  int o = (int)((i / Kpad) % O);
  int t = (int)(i / ((size_t)Kpad * O));
  float v = 0.f;
  if (c < I) {
    int ky = t / kh, kx = t % kh;
    v = w[(((size_t)o * I + c) * kh + ky) * kh + kx];
  }
  dst[i] = f2bf(v);
}

// ---------------------------------------------------------------------------
// Zero tail columns [from, ld) of a bf16 matrix.
// ---------------------------------------------------------------------------
__global__ void k_zerotail(ushort_t* __restrict__ dst, int ld, int from) {
  int nc = ld - from;
  size_t i = (size_t)blockIdx.x * blockDim.x + threadIdx.x;
  size_t tot = (size_t)P_ * nc;
  if (i >= tot) return;
  int c = from + (int)(i % nc);
  size_t p = i / nc;
  dst[p * ld + c] = 0;
}

// ---------------------------------------------------------------------------
// Implicit-GEMM conv via WMMA bf16.
// Block: 128 threads = 4 waves. M-tile = 32 pixels, N-tile = 64 channels
// (one 16-wide oc tile per wave, two M sub-tiles => 2 accumulators/wave,
// B fragment reused for both). The full 32 x ldA A-slab for the current tap
// is staged in LDS with gfx1250 async copies, so there is ONE barrier pair
// per tap with 2*ldA/32 WMMAs in between.
// Epilogue: BN (g/sqrt(1+eps)) or conv bias, optional ReLU; writes bf16 into
// up to two concat destinations and/or f32 NCHW output.
// ---------------------------------------------------------------------------
__global__ void k_gemm(const ushort_t* __restrict__ A, int ldA,
                       const ushort_t* __restrict__ Wt, int O, int taps,
                       const float* __restrict__ g, const float* __restrict__ bvec,
                       int relu,
                       ushort_t* __restrict__ dst1, int ld1, int co1,
                       ushort_t* __restrict__ dst2, int ld2, int co2,
                       float* __restrict__ outNCHW, int OC) {
  __shared__ __align__(16) ushort_t At[32 * MAX_LDA];

  const int tid  = threadIdx.x;
  const int wv   = tid >> 5;
  const int lane = tid & 31;
  const int pmBase = blockIdx.x * 32;
  const int ocTile = blockIdx.y * 4 + wv;
  const bool active = (ocTile * 16) < O;       // uniform per wave
  const int ocBase = ocTile * 16;

  // Cooperative-load role: 32 rows, 4 threads/row, chunk = ldA/4 elems each.
  const int row   = tid >> 2;
  const int seg   = tid & 3;
  const int chunk = ldA >> 2;                  // multiple of 8 (ldA % 32 == 0)
  const int prow  = pmBase + row;
  const int nIdx  = prow / HW_;
  const int hwIdx = prow % HW_;
  const int hIdx  = hwIdx / W_;
  const int wIdx  = hwIdx % W_;

  // LDS byte base of this thread's staging chunk.
  lds_ushort* lbase = (lds_ushort*)At;
  const unsigned ldsBase =
      (unsigned)(size_t)(lbase + row * ldA + seg * chunk);

  v8f acc0 = {};
  v8f acc1 = {};
  union V16 { uint4 q[2]; v16bf v; };

  const int m16 = lane & 15;
  const int kb  = (lane >> 4) * 8;             // A K-half select

  for (int t = 0; t < taps; ++t) {
    int dy = 0, dx = 0;
    if (taps == 9) { dy = t / 3 - 1; dx = t % 3 - 1; }
    const int hh = hIdx + dy, ww = wIdx + dx;
    const bool valid = (hh >= 0) && (hh < H_) && (ww >= 0) && (ww < W_);
    const ushort_t* srcRow =
        A + (size_t)(nIdx * HW_ + hh * W_ + ww) * ldA + seg * chunk;

    __syncthreads();                           // previous-tap reads done
    if (valid) {
      for (int e = 0; e < chunk; e += 8) {     // 8 bf16 = 16B per async op
        unsigned long long ga = (unsigned long long)(size_t)(srcRow + e);
        unsigned la = ldsBase + (unsigned)(e * 2);
        asm volatile("global_load_async_to_lds_b128 %0, %1, off"
                     :: "v"(la), "v"(ga) : "memory");
      }
    } else {
      uint4 z; z.x = 0u; z.y = 0u; z.z = 0u; z.w = 0u;
      for (int e = 0; e < chunk; e += 8)
        *(uint4*)&At[row * ldA + seg * chunk + e] = z;
    }
    asm volatile("s_wait_asynccnt 0x0" ::: "memory");
    __syncthreads();                           // slab visible to all waves

    if (active) {
      for (int k = 0; k < ldA; k += 32) {
        V16 a0, a1, ub;
        a0.q[0] = *(const uint4*)&At[m16 * ldA + k + kb];
        a0.q[1] = *(const uint4*)&At[m16 * ldA + k + kb + 16];
        a1.q[0] = *(const uint4*)&At[(16 + m16) * ldA + k + kb];
        a1.q[1] = *(const uint4*)&At[(16 + m16) * ldA + k + kb + 16];
        const ushort_t* bp =
            Wt + ((size_t)t * O + ocBase + (lane & 15)) * ldA + k + (lane >> 4) * 8;
        __builtin_prefetch(bp + 32, 0, 1);     // global_prefetch_b8
        ub.q[0] = ((const uint4*)bp)[0];
        ub.q[1] = *(const uint4*)(bp + 16);
        acc0 = __builtin_amdgcn_wmma_f32_16x16x32_bf16(
            false, a0.v, false, ub.v, (short)0, acc0, false, false);
        acc1 = __builtin_amdgcn_wmma_f32_16x16x32_bf16(
            false, a1.v, false, ub.v, (short)0, acc1, false, false);
      }
    }
  }

  if (!active) return;

  const int nOut = lane & 15;
  const int hi   = lane >> 4;
  const int oc   = ocBase + nOut;
  const float s  = g ? g[oc] * rsqrtf(1.0f + 1e-5f) : 1.0f;
  const float bb = bvec ? bvec[oc] : 0.0f;

#pragma unroll
  for (int half = 0; half < 2; ++half) {
#pragma unroll
    for (int vi = 0; vi < 8; ++vi) {
      const int m = half * 16 + vi + 8 * hi;
      const int p = pmBase + m;
      float val = (half ? acc1[vi] : acc0[vi]) * s + bb;
      if (relu) val = fmaxf(val, 0.0f);
      if (dst1) dst1[(size_t)p * ld1 + co1 + oc] = f2bf(val);
      if (dst2) dst2[(size_t)p * ld2 + co2 + oc] = f2bf(val);
      if (outNCHW) {
        int n = p / HW_, hw = p % HW_;
        outNCHW[((size_t)n * OC + oc) * HW_ + hw] = val;
      }
    }
  }
}

// ---------------------------------------------------------------------------
// Local 9x9 cross attention, one wave (32 lanes) per pixel, online softmax.
// Channels striped across lanes (c = j*32 + lane). Zero-padded borders match
// the reference (score contributes exp(0), value contributes nothing).
// ---------------------------------------------------------------------------
template <int NJ>
__global__ void k_attn(const ushort_t* __restrict__ Q,
                       const ushort_t* __restrict__ K,
                       const ushort_t* __restrict__ V,
                       int ldq, int C,
                       ushort_t* __restrict__ Out, int ldo) {
  const int wv = threadIdx.x >> 5;
  const int lane = threadIdx.x & 31;
  const int p = blockIdx.x * 4 + wv;           // P_ % 4 == 0
  const int n = p / HW_, hw = p % HW_;
  const int h = hw / W_, w = hw % W_;

  float q[NJ];
#pragma unroll
  for (int j = 0; j < NJ; ++j) {
    int c = j * 32 + lane;
    q[j] = (c < C) ? bf2f(Q[(size_t)p * ldq + c]) : 0.0f;
  }

  float m = -1e30f, d = 0.0f;
  float o[NJ];
#pragma unroll
  for (int j = 0; j < NJ; ++j) o[j] = 0.0f;

  for (int i = 0; i < 9; ++i) {
    for (int j2 = 0; j2 < 9; ++j2) {
      const int hh = h + i - 4, ww = w + j2 - 4;
      const bool valid = (hh >= 0) && (hh < H_) && (ww >= 0) && (ww < W_);
      const size_t pp = (size_t)(n * HW_ + hh * W_ + ww);
      float s = 0.0f;
      if (valid) {
        const ushort_t* kr = K + pp * ldq;
#pragma unroll
        for (int j = 0; j < NJ; ++j) {
          int c = j * 32 + lane;
          if (c < C) s += q[j] * bf2f(kr[c]);
        }
      }
#pragma unroll
      for (int off = 16; off >= 1; off >>= 1)
        s += __shfl_xor(s, off, 32);

      const float mn = fmaxf(m, s);
      const float fsc = __expf(m - mn);
      const float e   = __expf(s - mn);
      d = d * fsc + e;
#pragma unroll
      for (int j = 0; j < NJ; ++j) o[j] *= fsc;
      if (valid) {
        const ushort_t* vr = V + pp * ldq;
#pragma unroll
        for (int j = 0; j < NJ; ++j) {
          int c = j * 32 + lane;
          if (c < C) o[j] += e * bf2f(vr[c]);
        }
      }
      m = mn;
    }
  }

  const float inv = 1.0f / d;
#pragma unroll
  for (int j = 0; j < NJ; ++j) {
    int c = j * 32 + lane;
    if (c < C) Out[(size_t)p * ldo + c] = f2bf(o[j] * inv);
  }
}

// ---------------------------------------------------------------------------
// Host-side orchestration.
// ---------------------------------------------------------------------------
extern "C" void kernel_launch(void* const* d_in, const int* in_sizes, int n_in,
                              void* d_out, int out_size, void* d_ws, size_t ws_size,
                              hipStream_t stream) {
  (void)in_sizes; (void)n_in; (void)out_size; (void)ws_size;

  auto F = [&](int i) { return (const float*)d_in[i]; };

  // --- input index map (setup_inputs dict order, depth-first) ---
  const float* imgFeat   = F(0);   // (2,80,180,180)
  const float* lidarFeat = F(1);   // (2,256,180,180)
  const float* shImgW = F(2);  const float* shImgB = F(3);
  const float* shPtsW = F(4);  const float* shPtsB = F(5);
  const float* Iq1w=F(6),  *Iq1g=F(7),  *Iq1b=F(8);
  const float* Iq2w=F(9),  *Iq2g=F(10), *Iq2b=F(11);
  const float* Ik1w=F(12), *Ik1g=F(13), *Ik1b=F(14);
  const float* Ik2w=F(15), *Ik2g=F(16), *Ik2b=F(17);
  const float* Ivw =F(18), *Ivg =F(19), *Ivb =F(20);
  const float* Pq1w=F(21), *Pq1g=F(22), *Pq1b=F(23);
  const float* Pq2w=F(24), *Pq2g=F(25), *Pq2b=F(26);
  const float* Pk1w=F(27), *Pk1g=F(28), *Pk1b=F(29);
  const float* Pk2w=F(30), *Pk2g=F(31), *Pk2b=F(32);
  const float* Pvw =F(33), *Pvg =F(34), *Pvb =F(35);
  const float* IintW=F(36), *IintG=F(37), *IintB=F(38);
  const float* PintW=F(39), *PintG=F(40), *PintB=F(41);
  const float* finW=F(42),  *finG=F(43),  *finB=F(44);

  float* out = (float*)d_out;

  // --- workspace carve (bf16 = 2 bytes, 256B aligned) ---
  char* wp = (char*)d_ws;
  auto alloc = [&](size_t elems) -> ushort_t* {
    ushort_t* r = (ushort_t*)wp;
    wp += ((elems * 2 + 255) / 256) * 256;
    return r;
  };
  const int CI_P = 96, CP_P = 256, CATI = 160, CATP = 512, CATF = 352;
  ushort_t* rawImg = alloc((size_t)P_ * CI_P);
  ushort_t* rawPts = alloc((size_t)P_ * CP_P);
  ushort_t* imgMat = alloc((size_t)P_ * CI_P);
  ushort_t* ptsMat = alloc((size_t)P_ * CP_P);
  ushort_t* catI   = alloc((size_t)P_ * CATI);
  ushort_t* catP   = alloc((size_t)P_ * CATP);
  ushort_t* catF   = alloc((size_t)P_ * CATF);
  ushort_t* tmp96  = alloc((size_t)P_ * CI_P);
  ushort_t* q96    = alloc((size_t)P_ * CI_P);
  ushort_t* key96  = alloc((size_t)P_ * CI_P);
  ushort_t* val96  = alloc((size_t)P_ * CI_P);
  ushort_t* tmp256 = alloc((size_t)P_ * CP_P);
  ushort_t* q256   = alloc((size_t)P_ * CP_P);
  ushort_t* key256 = alloc((size_t)P_ * CP_P);
  ushort_t* val256 = alloc((size_t)P_ * CP_P);
  ushort_t* wShImg = alloc((size_t)9 * 80  * 96);
  ushort_t* wShPts = alloc((size_t)9 * 256 * 256);
  ushort_t* wIq1 = alloc((size_t)80 * 96);
  ushort_t* wIq2 = alloc((size_t)80 * 96);
  ushort_t* wIk1 = alloc((size_t)80 * 256);
  ushort_t* wIk2 = alloc((size_t)80 * 96);
  ushort_t* wIv  = alloc((size_t)80 * 256);
  ushort_t* wPq1 = alloc((size_t)256 * 256);
  ushort_t* wPq2 = alloc((size_t)256 * 256);
  ushort_t* wPk1 = alloc((size_t)256 * 96);
  ushort_t* wPk2 = alloc((size_t)256 * 256);
  ushort_t* wPv  = alloc((size_t)256 * 96);
  ushort_t* wIint = alloc((size_t)80 * 160);
  ushort_t* wPint = alloc((size_t)256 * 512);
  ushort_t* wFin  = alloc((size_t)9 * 256 * 352);

  auto gridFor = [](size_t tot) { return (unsigned)((tot + 255) / 256); };

  // --- activation & weight conversion ---
  k_tomat<<<gridFor((size_t)P_ * CI_P), 256, 0, stream>>>(imgFeat,   rawImg, 80,  CI_P);
  k_tomat<<<gridFor((size_t)P_ * CP_P), 256, 0, stream>>>(lidarFeat, rawPts, 256, CP_P);

  auto wc = [&](const float* src, ushort_t* dst, int O, int I, int kh, int Kpad) {
    size_t tot = (size_t)kh * kh * O * Kpad;
    k_wconv<<<gridFor(tot), 256, 0, stream>>>(src, dst, O, I, kh, Kpad);
  };
  wc(shImgW, wShImg, 80, 80, 3, 96);
  wc(shPtsW, wShPts, 256, 256, 3, 256);
  wc(Iq1w, wIq1, 80, 80, 1, 96);    wc(Iq2w, wIq2, 80, 80, 1, 96);
  wc(Ik1w, wIk1, 80, 256, 1, 256);  wc(Ik2w, wIk2, 80, 80, 1, 96);
  wc(Ivw,  wIv,  80, 256, 1, 256);
  wc(Pq1w, wPq1, 256, 256, 1, 256); wc(Pq2w, wPq2, 256, 256, 1, 256);
  wc(Pk1w, wPk1, 256, 80, 1, 96);   wc(Pk2w, wPk2, 256, 256, 1, 256);
  wc(Pvw,  wPv,  256, 80, 1, 96);
  wc(IintW, wIint, 80, 160, 1, 160);
  wc(PintW, wPint, 256, 512, 1, 512);
  wc(finW,  wFin,  256, 336, 3, 352);

  auto gemm = [&](const ushort_t* A, int ldA, const ushort_t* Wt, int O, int taps,
                  const float* g, const float* b, int relu,
                  ushort_t* d1, int l1, int c1,
                  ushort_t* d2, int l2, int c2,
                  float* of, int OC) {
    dim3 grid(P_ / 32, (unsigned)((O / 16 + 3) / 4));
    k_gemm<<<grid, 128, 0, stream>>>(A, ldA, Wt, O, taps, g, b, relu,
                                     d1, l1, c1, d2, l2, c2, of, OC);
  };

  // --- shared 3x3 convs (conv bias, no BN/ReLU); dual-write into concat ---
  gemm(rawImg, CI_P, wShImg, 80, 9, nullptr, shImgB, 0,
       imgMat, CI_P, 0, catI, CATI, 80, nullptr, 0);
  gemm(rawPts, CP_P, wShPts, 256, 9, nullptr, shPtsB, 0,
       ptsMat, CP_P, 0, catP, CATP, 256, nullptr, 0);

  // --- I_IML projections (ConvBNReLU chains) ---
  gemm(imgMat, CI_P, wIq1, 80, 1, Iq1g, Iq1b, 1, tmp96, CI_P, 0, nullptr,0,0, nullptr,0);
  gemm(tmp96,  CI_P, wIq2, 80, 1, Iq2g, Iq2b, 1, q96,   CI_P, 0, nullptr,0,0, nullptr,0);
  gemm(ptsMat, CP_P, wIk1, 80, 1, Ik1g, Ik1b, 1, tmp96, CI_P, 0, nullptr,0,0, nullptr,0);
  gemm(tmp96,  CI_P, wIk2, 80, 1, Ik2g, Ik2b, 1, key96, CI_P, 0, nullptr,0,0, nullptr,0);
  gemm(ptsMat, CP_P, wIv,  80, 1, Ivg,  Ivb,  1, val96, CI_P, 0, nullptr,0,0, nullptr,0);
  k_attn<3><<<P_ / 4, 128, 0, stream>>>(q96, key96, val96, CI_P, 80, catI, CATI);
  gemm(catI, CATI, wIint, 80, 1, IintG, IintB, 0, catF, CATF, 0, nullptr,0,0, nullptr,0);

  // --- P_IML projections ---
  gemm(ptsMat, CP_P, wPq1, 256, 1, Pq1g, Pq1b, 1, tmp256, CP_P, 0, nullptr,0,0, nullptr,0);
  gemm(tmp256, CP_P, wPq2, 256, 1, Pq2g, Pq2b, 1, q256,   CP_P, 0, nullptr,0,0, nullptr,0);
  gemm(imgMat, CI_P, wPk1, 256, 1, Pk1g, Pk1b, 1, tmp256, CP_P, 0, nullptr,0,0, nullptr,0);
  gemm(tmp256, CP_P, wPk2, 256, 1, Pk2g, Pk2b, 1, key256, CP_P, 0, nullptr,0,0, nullptr,0);
  gemm(imgMat, CI_P, wPv,  256, 1, Pvg,  Pvb,  1, val256, CP_P, 0, nullptr,0,0, nullptr,0);
  k_attn<8><<<P_ / 4, 128, 0, stream>>>(q256, key256, val256, CP_P, 256, catP, CATP);
  gemm(catP, CATP, wPint, 256, 1, PintG, PintB, 0, catF, CATF, 80, nullptr,0,0, nullptr,0);

  // --- final 3x3 conv over concat(new_img, new_pts) + BN + ReLU -> f32 NCHW ---
  k_zerotail<<<gridFor((size_t)P_ * (CATF - 336)), 256, 0, stream>>>(catF, CATF, 336);
  gemm(catF, CATF, wFin, 256, 9, finG, finB, 1,
       nullptr, 0, 0, nullptr, 0, 0, out, 256);
}
